// Model_51101520888402
// MI455X (gfx1250) — matrix operand order repaired and checked by reading, not compiled, not run on hardware
//
#include <hip/hip_runtime.h>
#include <hip/hip_bf16.h>

#define B_   512
#define L_   336
#define C_   321
#define WIN_ 12
#define S_   28
#define HID_ 25
#define PSEG_ 8

typedef __attribute__((ext_vector_type(16))) __bf16 v16bf;
typedef __attribute__((ext_vector_type(8)))  __bf16 v8bf;
typedef __attribute__((ext_vector_type(8)))  float  v8f;

union ABf { v16bf v; v8bf h[2]; };

// gfx1250 V_TANH_F32 (1 trans op); fallback to exp2+rcp otherwise.
#if __has_builtin(__builtin_amdgcn_tanhf)
__device__ __forceinline__ float ftanh(float x) { return __builtin_amdgcn_tanhf(x); }
#elif __has_builtin(__builtin_amdgcn_tanh_f32)
__device__ __forceinline__ float ftanh(float x) { return __builtin_amdgcn_tanh_f32(x); }
#else
__device__ __forceinline__ float ftanh(float x) {
  float e = __builtin_amdgcn_exp2f(-2.885390f * x);  // exp(-2x)
  return __builtin_fmaf(-2.0f, __builtin_amdgcn_rcpf(1.0f + e), 1.0f);
}
#endif
// sigmoid of (2*x) given pre-scaled argument x = 0.5*gate  (0.5 folded into W)
__device__ __forceinline__ float fsig_pre(float xh) {
  return __builtin_fmaf(ftanh(xh), 0.5f, 0.5f);
}

// ---------------- Kernel 1: mean + windowed segment projection -------------
__global__ void k_segment(const float* __restrict__ x,
                          const float* __restrict__ seg,
                          float* __restrict__ xseg,
                          float* __restrict__ meanb) {
  int c = blockIdx.x * 64 + threadIdx.x;
  int b = blockIdx.y;
  if (c >= C_) return;
  float sr[WIN_]; float ssum = 0.f;
  #pragma unroll
  for (int w = 0; w < WIN_; ++w) { sr[w] = seg[c * WIN_ + w]; ssum += sr[w]; }
  float wp[S_]; float tot = 0.f;
  const float* xb = x + (size_t)b * L_ * C_ + c;
  for (int s = 0; s < S_; ++s) {
    float a = 0.f;
    #pragma unroll
    for (int w = 0; w < WIN_; ++w) {
      float v = xb[(size_t)(s * WIN_ + w) * C_];
      a += v * sr[w];
      tot += v;
    }
    wp[s] = a;
  }
  float mean = tot * (1.0f / (float)L_);
  meanb[b * C_ + c] = mean;
  for (int s = 0; s < S_; ++s)
    xseg[((size_t)c * S_ + s) * B_ + b] = wp[s] - mean * ssum;
}

// ---------------- Kernel 2: per-channel LSTM encoder+decoder ---------------
// One workgroup per channel; 8 waves x 4 batch-tiles of 16 rows = 512 batches.
// Gates remapped to 4x32 padded columns (8 WMMA N-tiles); tile pair (2k,2k+1)
// holds gate k so i/f/g/o align per-lane across accumulators.
// K-padding reuse: A col 25 = xt, col 26 = 1.0; B row 25 = Wx, row 26 = bias.
// i/f/o gate columns pre-scaled by 0.5 so sigmoid = tanh + fma (no mul).

#define CELL(q, mt) do {                                                       \
  ABf ua;                                                                      \
  ua.h[0] = *reinterpret_cast<const v8bf*>(&hbuf[mt][row16][hi * 8]);          \
  ua.h[1] = *reinterpret_cast<const v8bf*>(&hbuf[mt][row16][16 + hi * 8]);     \
  v8f acc[8];                                                                  \
  _Pragma("unroll") for (int t = 0; t < 8; ++t) {                              \
    v8f cz = {};                                                               \
    acc[t] = __builtin_amdgcn_wmma_f32_16x16x32_bf16(                          \
        false, ua.v, false, bfr[t], (short)0, cz, false, false);               \
  }                                                                            \
  _Pragma("unroll") for (int ss = 0; ss < 2; ++ss) {                           \
    _Pragma("unroll") for (int r = 0; r < 8; ++r) {                            \
      float sf = fsig_pre(acc[2 + ss][r]);                                     \
      float si = fsig_pre(acc[0 + ss][r]);                                     \
      float tg = ftanh(acc[4 + ss][r]);                                        \
      float so = fsig_pre(acc[6 + ss][r]);                                     \
      float cn = __builtin_fmaf(sf, cst[q][ss][r], si * tg);                   \
      float hn = so * ftanh(cn);                                               \
      cst[q][ss][r] = cn;                                                      \
      if (ss == 0 || row16 < 9) /* keep xt/1.0/zero K-columns intact */        \
        hbuf[mt][8 * hi + r][16 * ss + row16] = (__bf16)hn;                    \
    }                                                                          \
  }                                                                            \
} while (0)

// Projection h -> scalar; duplicate (idempotent) across half-waves, no branch.
// Also feeds decoder input back into A col 25.
#define PROJ(mt, p) do {                                                       \
  float sum = bpc;                                                             \
  _Pragma("unroll") for (int j = 0; j < HID_; ++j)                             \
    sum += (float)hbuf[mt][row16][j] * wpb[j];                                 \
  preds[((size_t)c0 * PSEG_ + (p)) * B_ + (mt) * 16 + row16] = sum;            \
  hbuf[mt][row16][HID_] = (__bf16)sum;                                         \
} while (0)

__global__ void __launch_bounds__(256, 1) k_lstm(
    const float* __restrict__ xseg, const float* __restrict__ W_ih,
    const float* __restrict__ W_hh, const float* __restrict__ b_ih,
    const float* __restrict__ b_hh, const float* __restrict__ Wp,
    const float* __restrict__ bp, float* __restrict__ preds) {
  __shared__ __align__(16) __bf16 Wb[8][16][32];    // B frags: [tile][n][k]
  __shared__ __align__(16) __bf16 hbuf[32][16][32]; // [mtile][m][k]: h|xt|1|0
  __shared__ float wpb[32];

  const int c0 = blockIdx.x;
  const int tid = threadIdx.x;

  // ---- stage padded weights: col = 16*t + n -> gate kg = col/32, j = col%32
  //      k<25: W_hh; k==25: Wx; k==26: b_ih+b_hh; else 0.
  //      Gates i(kg=0), f(1), o(3) scaled by 0.5 (sigmoid via tanh).
  for (int idx = tid; idx < 8 * 16 * 32; idx += 256) {
    int t = idx >> 9, n = (idx >> 5) & 15, k = idx & 31;
    int col = 16 * t + n, kg = col >> 5, j = col & 31;
    float v = 0.f;
    if (j < HID_) {
      size_t g = (size_t)c0 * 100 + 25 * kg + j;
      if (k < HID_)            v = W_hh[g * HID_ + k];
      else if (k == HID_)      v = W_ih[g];
      else if (k == HID_ + 1)  v = b_ih[g] + b_hh[g];
      if (kg != 2) v *= 0.5f;  // pre-scale sigmoid gates
    }
    Wb[t][n][k] = (__bf16)v;
  }
  for (int j = tid; j < 32; j += 256)
    wpb[j] = (j < HID_) ? Wp[(size_t)c0 * HID_ + j] : 0.f;
  // h0 = 0; constant-1 column at k = 26
  for (int idx = tid; idx < 32 * 16 * 32; idx += 256)
    ((__bf16*)hbuf)[idx] = (__bf16)(((idx & 31) == HID_ + 1) ? 1.0f : 0.0f);
  __syncthreads();

  const float bpc = bp[c0];
  const int lane = tid & 31;
  const int wave = tid >> 5;
  const int row16 = lane & 15;
  const int hi = lane >> 4;       // half-wave select (A/B fragment K halves)
  const int mtb = wave * 4;       // this wave's first batch tile

  // persistent B fragments in VGPRs (8 tiles x 8 VGPRs)
  v16bf bfr[8];
  #pragma unroll
  for (int t = 0; t < 8; ++t) {
    ABf u;
    u.h[0] = *reinterpret_cast<const v8bf*>(&Wb[t][row16][hi * 8]);
    u.h[1] = *reinterpret_cast<const v8bf*>(&Wb[t][row16][16 + hi * 8]);
    bfr[t] = u.v;
  }

  float cst[4][2][8];
  #pragma unroll
  for (int q = 0; q < 4; ++q)
    #pragma unroll
    for (int s = 0; s < 2; ++s)
      #pragma unroll
      for (int r = 0; r < 8; ++r) cst[q][s][r] = 0.f;

  // ---- encoder: 28 steps, xt fetch software-pipelined one step ahead so the
  //      global-load wait overlaps a full step of WMMA + activations.
  float xn[4];
  #pragma unroll
  for (int q = 0; q < 4; ++q)
    xn[q] = xseg[(size_t)c0 * S_ * B_ + (mtb + q) * 16 + row16];
  for (int s = 0; s < S_; ++s) {
    #pragma unroll
    for (int q = 0; q < 4; ++q)
      hbuf[mtb + q][row16][HID_] = (__bf16)xn[q];  // idempotent across halves
    int sn = (s + 1 < S_) ? s + 1 : s;             // prefetch next step
    #pragma unroll
    for (int q = 0; q < 4; ++q)
      xn[q] = xseg[((size_t)c0 * S_ + sn) * B_ + (mtb + q) * 16 + row16];
    #pragma unroll
    for (int q = 0; q < 4; ++q) CELL(q, mtb + q);
  }

  // ---- projection of final encoder state -> preds[:,0] + decoder input
  #pragma unroll
  for (int q = 0; q < 4; ++q) PROJ(mtb + q, 0);

  // ---- decoder: 7 steps; xt slot already holds previous projection
  for (int p = 1; p < PSEG_; ++p) {
    #pragma unroll
    for (int q = 0; q < 4; ++q) {
      const int mt = mtb + q;
      CELL(q, mt);
      PROJ(mt, p);
    }
  }
}

// ---------------- Kernel 3: expand preds through window + add mean ---------
__global__ void k_expand(const float* __restrict__ preds,
                         const float* __restrict__ seg,
                         const float* __restrict__ meanb,
                         float* __restrict__ out) {
  size_t idx = (size_t)blockIdx.x * blockDim.x + threadIdx.x;
  const size_t N = (size_t)B_ * (PSEG_ * WIN_) * C_;
  if (idx >= N) return;
  int c = (int)(idx % C_);
  int t = (int)((idx / C_) % (PSEG_ * WIN_));
  int b = (int)(idx / ((size_t)C_ * (PSEG_ * WIN_)));
  int p = t / WIN_, w = t % WIN_;
  out[idx] = preds[((size_t)c * PSEG_ + p) * B_ + b] * seg[c * WIN_ + w] +
             meanb[b * C_ + c];
}

extern "C" void kernel_launch(void* const* d_in, const int* in_sizes, int n_in,
                              void* d_out, int out_size, void* d_ws, size_t ws_size,
                              hipStream_t stream) {
  const float* x    = (const float*)d_in[0];
  const float* seg  = (const float*)d_in[1];  // (C,12,1) contiguous == (C,12)
  const float* W_ih = (const float*)d_in[2];  // (C,100,1)
  const float* W_hh = (const float*)d_in[3];  // (C,100,25)
  const float* b_ih = (const float*)d_in[4];
  const float* b_hh = (const float*)d_in[5];
  const float* Wp   = (const float*)d_in[6];  // (C,1,25)
  const float* bp   = (const float*)d_in[7];  // (C,1)
  float* out = (float*)d_out;

  float* xseg  = (float*)d_ws;                        // C*S*B
  float* meanb = xseg + (size_t)C_ * S_ * B_;         // B*C
  float* preds = meanb + (size_t)B_ * C_;             // C*PSEG*B

  dim3 g1((C_ + 63) / 64, B_);
  k_segment<<<g1, 64, 0, stream>>>(x, seg, xseg, meanb);

  k_lstm<<<C_, 256, 0, stream>>>(xseg, W_ih, W_hh, b_ih, b_hh, Wp, bp, preds);

  const size_t N = (size_t)B_ * (PSEG_ * WIN_) * C_;
  k_expand<<<(unsigned)((N + 255) / 256), 256, 0, stream>>>(preds, seg, meanb, out);
}
